// RoPEMaskedAttentionHead_1529008357804
// MI455X (gfx1250) — matrix operand order repaired and checked by reading, not compiled, not run on hardware
//
#include <hip/hip_runtime.h>

// ---------------------------------------------------------------------------
// RoPE + causal attention head for MI455X (gfx1250), bf16 WMMA pipeline.
//   b=4, m=2048, d=1024, fp32 in/out.
// Pipeline: cast->bf16, fused QKV NT-GEMM (WMMA), RoPE in-place on Q/K,
// flash-attention (Q staged in LDS, parallel online softmax, WMMA S and P*V).
// ---------------------------------------------------------------------------

#define BSZ  4
#define MLEN 2048
#define DDIM 1024
#define CH   256          // key chunk per attention iteration
#define SSTR 257          // f32 LDS row stride (bank-conflict pad)
#define PSTR 264          // bf16 LDS row stride (16B-aligned, conflict pad)
#define QSTR 1032         // bf16 LDS row stride for Q tile (16B-aligned pad)

typedef __attribute__((ext_vector_type(16))) __bf16 v16bf;
typedef __attribute__((ext_vector_type(8)))  __bf16 v8bf;
typedef __attribute__((ext_vector_type(8)))  float  v8f;

__device__ __forceinline__ v8bf ld8(const __bf16* p) {
    return *reinterpret_cast<const v8bf*>(p);
}

// A-matrix fragment 16x32 (MxK) bf16, wave32 layout:
// lanes 0-15: row M=lane, K {k0..k0+7, k0+16..k0+23}
// lanes16-31: row M=lane-16, K {k0+8..k0+15, k0+24..k0+31}
__device__ __forceinline__ v16bf load_afrag(const __bf16* base, int ldm,
                                            int row0, int k0, int lane) {
    const int half = lane >> 4;
    const __bf16* r = base + (size_t)(row0 + (lane & 15)) * ldm;
    v8bf lo = ld8(r + k0 + half * 8);
    v8bf hi = ld8(r + k0 + 16 + half * 8);
    v16bf a;
#pragma unroll
    for (int i = 0; i < 8; ++i) { a[i] = lo[i]; a[i + 8] = hi[i]; }
    return a;
}

// B-matrix fragment 32x16 (KxN) bf16 where B[K][N] = src[n0+N][k0+K]
// (src row-major over N rows, K contiguous -> NT gemm, contiguous loads):
// lanes 0-15: N=lane,    K {k0..k0+7, k0+16..k0+23}
// lanes16-31: N=lane-16, K {k0+8..k0+15, k0+24..k0+31}
__device__ __forceinline__ v16bf load_bfrag(const __bf16* base, int ldm,
                                            int n0, int k0, int lane) {
    const int half = lane >> 4;
    const __bf16* r = base + (size_t)(n0 + (lane & 15)) * ldm + k0 + half * 8;
    v8bf lo = ld8(r);
    v8bf hi = ld8(r + 16);
    v16bf b;
#pragma unroll
    for (int i = 0; i < 8; ++i) { b[i] = lo[i]; b[i + 8] = hi[i]; }
    return b;
}

__device__ __forceinline__ v8f wmma_bf16(v16bf a, v16bf b, v8f c) {
    // (neg_a, A, neg_b, B, c_mod, C, reuse_a, reuse_b)
    return __builtin_amdgcn_wmma_f32_16x16x32_bf16(false, a, false, b,
                                                   (short)0, c, false, false);
}

// ---------------------------------------------------------------------------
// fp32 -> bf16 cast
// ---------------------------------------------------------------------------
__global__ void cast_bf16_kernel(const float* __restrict__ src,
                                 __bf16* __restrict__ dst, size_t n) {
    size_t i = (size_t)blockIdx.x * blockDim.x + threadIdx.x;
    if (i < n) dst[i] = (__bf16)src[i];
}

// ---------------------------------------------------------------------------
// Fused QKV projection: q[bm,e] = sum_d x[bm,d] * w[e,d]  (NT gemm).
// grid (DDIM/256, (B*M)/16, 3), block 128 (4 waves, each owns 64 cols).
// z=0 -> Q (bf16 [b*m][d]), z=1 -> K, z=2 -> V stored TRANSPOSED [b][d][m].
// ---------------------------------------------------------------------------
__global__ __launch_bounds__(128) void gemm_qkv_kernel(
    const __bf16* __restrict__ xb,
    const __bf16* __restrict__ wq, const __bf16* __restrict__ wk,
    const __bf16* __restrict__ wv,
    __bf16* __restrict__ qb, __bf16* __restrict__ kb,
    __bf16* __restrict__ vbt) {
    const int lane = threadIdx.x & 31;
    const int wave = threadIdx.x >> 5;
    const int half = lane >> 4;
    const int l15  = lane & 15;
    const int n0   = blockIdx.x * 256 + wave * 64;
    const int row0 = blockIdx.y * 16;           // row in flattened [B*M]
    const int z    = blockIdx.z;

    const __bf16* w = (z == 0) ? wq : ((z == 1) ? wk : wv);

    v8f acc[4] = {};
    for (int k0 = 0; k0 < DDIM; k0 += 32) {
        v16bf a = load_afrag(xb, DDIM, row0, k0, lane);
#pragma unroll
        for (int nt = 0; nt < 4; ++nt) {
            v16bf b = load_bfrag(w, DDIM, n0 + nt * 16, k0, lane);
            acc[nt] = wmma_bf16(a, b, acc[nt]);
        }
    }

    if (z < 2) {
        __bf16* dst = (z == 0) ? qb : kb;
#pragma unroll
        for (int nt = 0; nt < 4; ++nt)
#pragma unroll
            for (int j = 0; j < 8; ++j) {
                int row = half * 8 + j;
                dst[(size_t)(row0 + row) * DDIM + n0 + nt * 16 + l15] =
                    (__bf16)acc[nt][j];
            }
    } else {
        const int bidx = row0 / MLEN;
        const int mm   = row0 % MLEN;
#pragma unroll
        for (int nt = 0; nt < 4; ++nt)
#pragma unroll
            for (int j = 0; j < 8; ++j) {
                int row = half * 8 + j;   // m offset
                int e   = n0 + nt * 16 + l15;
                vbt[((size_t)bidx * DDIM + e) * MLEN + mm + row] =
                    (__bf16)acc[nt][j];
            }
    }
}

// ---------------------------------------------------------------------------
// RoPE applied in place to Q and K (bf16). One thread per (b, m, pair i).
// theta_i = 10000^(-2*(i-1)/d)  (matches reference's (i-1) quirk)
// ---------------------------------------------------------------------------
__global__ void rope_kernel(__bf16* __restrict__ qb, __bf16* __restrict__ kb) {
    const size_t total = (size_t)BSZ * MLEN * (DDIM / 2);
    size_t idx = (size_t)blockIdx.x * blockDim.x + threadIdx.x;
    if (idx >= total) return;
    int    i    = (int)(idx % (DDIM / 2));
    size_t bm   = idx / (DDIM / 2);
    int    mrow = (int)(bm % MLEN);

    // ln(10000) = 9.210340371976184
    float theta = __expf(-2.0f * ((float)i - 1.0f) * (9.210340371976184f / (float)DDIM));
    float ang   = (float)mrow * theta;
    float s, c;
    __sincosf(ang, &s, &c);

    size_t off = bm * DDIM + 2 * (size_t)i;
    {
        float te = (float)qb[off], to = (float)qb[off + 1];
        qb[off]     = (__bf16)(te * c + to * s);
        qb[off + 1] = (__bf16)(-te * s + to * c);
    }
    {
        float te = (float)kb[off], to = (float)kb[off + 1];
        kb[off]     = (__bf16)(te * c + to * s);
        kb[off + 1] = (__bf16)(-te * s + to * c);
    }
}

// ---------------------------------------------------------------------------
// Flash attention with causal mask. grid (MLEN/16, BSZ), block 256 (8 waves).
// Q tile (16x1024 bf16) staged once in LDS. Per key-chunk of 256:
//   Phase A : each wave computes 2 S tiles (16x16, 32 WMMAs over d) -> LDS f32
//   Phase B1: ALL 8 waves do online-softmax; wave w owns rows {2w, 2w+1}
//             (one per 16-lane half); lane scans 16 cols; shfl_xor reduction.
//   Phase B2: all waves rescale their O slice and do P*V (WMMA), each wave
//             owning 128 output columns (V read from transposed layout).
// ---------------------------------------------------------------------------
__global__ __launch_bounds__(256) void attn_kernel(
    const __bf16* __restrict__ qb, const __bf16* __restrict__ kb,
    const __bf16* __restrict__ vbt, float* __restrict__ out) {
    __shared__ __bf16 Q_lds[16 * QSTR];     // staged Q tile
    __shared__ float  S_lds[16 * SSTR];
    __shared__ __bf16 P_lds[16 * PSTR];
    __shared__ float  A_lds[16];
    __shared__ float  L_lds[16];

    const int b    = blockIdx.y;
    const int q0   = blockIdx.x * 16;
    const int tid  = threadIdx.x;
    const int lane = tid & 31;
    const int wave = tid >> 5;
    const int half = lane >> 4;
    const int l15  = lane & 15;
    const int n0w  = wave * 128;            // this wave's output column slice

    const __bf16* q_base = qb + (size_t)b * MLEN * DDIM;
    const __bf16* k_base = kb + (size_t)b * MLEN * DDIM;
    const __bf16* v_base = vbt + (size_t)b * DDIM * MLEN;

    // ---- Stage the Q tile into LDS (once per block) ----------------------
    // 16 rows x 1024 halfs = 2048 chunks of 8 halfs; 8 chunks per thread.
#pragma unroll
    for (int t = 0; t < 8; ++t) {
        int c   = tid + t * 256;            // chunk index
        int row = c >> 7;                   // 128 chunks per row
        int col = (c & 127) * 8;
        *reinterpret_cast<v8bf*>(&Q_lds[row * QSTR + col]) =
            ld8(q_base + (size_t)(q0 + row) * DDIM + col);
    }
    __syncthreads();

    v8f oacc[8] = {};
    // Online-softmax state: wave w / half h owns row (2w + h); replicated
    // across the 16 lanes of that half.
    const int myrow = wave * 2 + half;
    float m_run = -__builtin_inff();
    float l_run = 0.0f;

    const float scale = 0.03125f;           // 1/sqrt(1024)
    const int nch = (q0 + 16 + CH - 1) / CH;

    for (int ch = 0; ch < nch; ++ch) {
        const int kbeg = ch * CH;

        // ---- Phase A: score tiles -> S_lds -------------------------------
#pragma unroll
        for (int t = 0; t < 2; ++t) {
            const int ktile = wave * 2 + t;
            const int k0    = kbeg + ktile * 16;
            if (k0 <= q0 + 15) {            // wave-uniform branch (EXEC safe)
                v8f s = {};
                for (int kk = 0; kk < DDIM; kk += 32) {
                    v16bf a  = load_afrag(Q_lds, QSTR, 0, kk, lane);
                    v16bf bb = load_bfrag(k_base, DDIM, k0, kk, lane);
                    s = wmma_bf16(a, bb, s);
                }
#pragma unroll
                for (int j = 0; j < 8; ++j) {
                    int row = half * 8 + j;
                    float v = s[j] * scale;
                    if (k0 + l15 > q0 + row) v = -__builtin_inff();
                    S_lds[row * SSTR + ktile * 16 + l15] = v;
                }
            } else {
#pragma unroll
                for (int j = 0; j < 8; ++j) {
                    int row = half * 8 + j;
                    S_lds[row * SSTR + ktile * 16 + l15] = -__builtin_inff();
                }
            }
        }
        // Prefetch next chunk's K rows while S is consumed (gfx1250 path).
        if (kbeg + CH <= q0 + 15) {
            __builtin_prefetch(k_base + (size_t)(kbeg + CH + wave * 2 * 16 + l15) * DDIM,
                               0, 1);
        }
        __syncthreads();

        // ---- Phase B1: parallel online softmax + P -----------------------
        {
            const int r  = myrow;
            const int c0 = l15 * 16;        // this lane's 16-column slice
            float mc = -__builtin_inff();
#pragma unroll
            for (int c = 0; c < 16; ++c)
                mc = fmaxf(mc, S_lds[r * SSTR + c0 + c]);
            // reduce max across the 16 lanes of this half (wave32-safe)
#pragma unroll
            for (int msk = 1; msk < 16; msk <<= 1)
                mc = fmaxf(mc, __shfl_xor(mc, msk, 32));

            float mnew  = fmaxf(m_run, mc);
            float alpha = __expf(m_run - mnew);    // exp(-inf)=0 on chunk 0
            float psum  = 0.0f;
#pragma unroll
            for (int c = 0; c < 16; ++c) {
                float p = __expf(S_lds[r * SSTR + c0 + c] - mnew);
                P_lds[r * PSTR + c0 + c] = (__bf16)p;
                psum += p;
            }
#pragma unroll
            for (int msk = 1; msk < 16; msk <<= 1)
                psum += __shfl_xor(psum, msk, 32);

            l_run = l_run * alpha + psum;
            m_run = mnew;
            if (l15 == 0) A_lds[r] = alpha;
        }
        __syncthreads();

        // ---- Phase B2: rescale O, then O += P * V (WMMA) -----------------
        float al[8];
#pragma unroll
        for (int j = 0; j < 8; ++j) al[j] = A_lds[half * 8 + j];
#pragma unroll
        for (int nt = 0; nt < 8; ++nt)
#pragma unroll
            for (int j = 0; j < 8; ++j) oacc[nt][j] *= al[j];

        for (int ks = 0; ks < CH; ks += 32) {
            v16bf a = load_afrag(P_lds, PSTR, 0, ks, lane);
            // prefetch V one K-step ahead
            __builtin_prefetch(v_base + (size_t)(n0w + l15 * 8) * MLEN +
                                   kbeg + ks + 32, 0, 1);
#pragma unroll
            for (int nt = 0; nt < 8; ++nt) {
                v16bf bb = load_bfrag(v_base, MLEN, n0w + nt * 16,
                                      kbeg + ks, lane);
                oacc[nt] = wmma_bf16(a, bb, oacc[nt]);
            }
        }
        __syncthreads();
    }

    if (l15 == 0) L_lds[myrow] = l_run;
    __syncthreads();

    float linv[8];
#pragma unroll
    for (int j = 0; j < 8; ++j) linv[j] = 1.0f / L_lds[half * 8 + j];

    float* ob = out + ((size_t)b * MLEN + q0) * DDIM;
#pragma unroll
    for (int nt = 0; nt < 8; ++nt)
#pragma unroll
        for (int j = 0; j < 8; ++j) {
            int row = half * 8 + j;
            ob[(size_t)row * DDIM + n0w + nt * 16 + l15] = oacc[nt][j] * linv[j];
        }
}

// ---------------------------------------------------------------------------
// Host launcher
// ---------------------------------------------------------------------------
extern "C" void kernel_launch(void* const* d_in, const int* in_sizes, int n_in,
                              void* d_out, int out_size, void* d_ws,
                              size_t ws_size, hipStream_t stream) {
    (void)in_sizes; (void)n_in; (void)out_size; (void)ws_size;
    const float* x   = (const float*)d_in[0];
    const float* w_q = (const float*)d_in[1];
    const float* w_k = (const float*)d_in[2];
    const float* w_v = (const float*)d_in[3];
    float* out = (float*)d_out;

    const size_t NX = (size_t)BSZ * MLEN * DDIM;   // 8M elems
    const size_t NW = (size_t)DDIM * DDIM;         // 1M elems

    // Workspace layout (bf16 buffers), all offsets 256B aligned.
    char* ws = (char*)d_ws;
    size_t off = 0;
    auto alloc = [&](size_t bytes) {
        void* p = ws + off;
        off += (bytes + 255) & ~(size_t)255;
        return p;
    };
    __bf16* xb  = (__bf16*)alloc(NX * 2);          // x, bf16 [b*m][d]
    __bf16* wqb = (__bf16*)alloc(NW * 2);
    __bf16* wkb = (__bf16*)alloc(NW * 2);
    __bf16* wvb = (__bf16*)alloc(NW * 2);
    __bf16* qb  = (__bf16*)alloc(NX * 2);          // Q (roped in place)
    __bf16* kb  = (__bf16*)alloc(NX * 2);          // K (roped in place)
    __bf16* vbt = (__bf16*)alloc(NX * 2);          // V transposed [b][d][m]

    // 1) casts to bf16
    cast_bf16_kernel<<<(unsigned)((NX + 255) / 256), 256, 0, stream>>>(x, xb, NX);
    cast_bf16_kernel<<<(unsigned)((NW + 255) / 256), 256, 0, stream>>>(w_q, wqb, NW);
    cast_bf16_kernel<<<(unsigned)((NW + 255) / 256), 256, 0, stream>>>(w_k, wkb, NW);
    cast_bf16_kernel<<<(unsigned)((NW + 255) / 256), 256, 0, stream>>>(w_v, wvb, NW);

    // 2) fused QKV projections (WMMA)
    gemm_qkv_kernel<<<dim3(DDIM / 256, (BSZ * MLEN) / 16, 3), 128, 0, stream>>>(
        xb, wqb, wkb, wvb, qb, kb, vbt);

    // 3) RoPE in place on Q and K
    const size_t NR = (size_t)BSZ * MLEN * (DDIM / 2);
    rope_kernel<<<(unsigned)((NR + 255) / 256), 256, 0, stream>>>(qb, kb);

    // 4) causal flash attention (WMMA for S and P*V)
    attn_kernel<<<dim3(MLEN / 16, BSZ), 256, 0, stream>>>(qb, kb, vbt, out);
}
